// Quantize_24910810316943
// MI455X (gfx1250) — matrix operand order repaired and checked by reading, not compiled, not run on hardware
//
#include <hip/hip_runtime.h>
#include <hip/hip_bf16.h>
#include <math.h>

typedef __attribute__((ext_vector_type(16))) _Float16 v16h;
typedef __attribute__((ext_vector_type(8)))  _Float16 v8h;
typedef __attribute__((ext_vector_type(8)))  float    v8f;

// Problem constants
#define DIMC   256          // channels / K
#define NE     1024         // codebook size
#define NB     32           // batch
#define HW     1024         // 32*32 pixels per image
#define NPIX   (NB*HW)      // 32768 total vectors
#define OUTN   (NB*DIMC*HW) // 8388608 floats of main output

// LDS A-tile padded row stride (multiple of 8 halves -> 16B aligned b128 reads)
#define ASTRIDE 264
#define MTILE   64          // pixels per block (4 waves * 16 rows)

// ---------------------------------------------------------------------------
// Kernel 1: embed[256][1024] f32 -> embedT[1024][256] f16 (+ per-code norms,
// zero histogram). One block per code j.
// ---------------------------------------------------------------------------
__global__ void __launch_bounds__(256)
prep_kernel(const float* __restrict__ embed,
            _Float16* __restrict__ embedT,
            float* __restrict__ norms,
            int* __restrict__ counts) {
    const int j = blockIdx.x;      // code
    const int c = threadIdx.x;     // channel
    float v = embed[(size_t)c * NE + j];
    embedT[(size_t)j * DIMC + c] = (_Float16)v;

    float s = v * v;
    for (int m = 16; m >= 1; m >>= 1) s += __shfl_xor(s, m, 32);
    __shared__ float ls[8];
    const int wave = c >> 5, lane = c & 31;
    if (lane == 0) ls[wave] = s;
    __syncthreads();
    if (c == 0) {
        float t = 0.f;
        for (int w = 0; w < 8; ++w) t += ls[w];
        norms[j] = t;
        counts[j] = 0;
    }
}

// ---------------------------------------------------------------------------
// Kernel 2 (fused): WMMA distance GEMM + argmin + codebook gather + diff.
// Grid: 512 blocks (32 images * 16 pixel-tiles of 64), 128 threads (4 waves).
//  - Each wave owns 16 pixel rows; its 16x256 f16 A block stays in VGPRs.
//  - B tiles (16 codes x 256 K, 8KB) staged in LDS double-buffered, shared by
//    all waves; next tile prefetched to regs during the WMMA chain.
//  - ||x_p||^2 accumulated in f32 during staging (p = tid&63 is invariant).
//  - diff element = ||x||^2 + (norm_j - 2 dot) = xnorm + winning minV.
//  - Output written directly from the original f32 embed (no x re-read).
// ---------------------------------------------------------------------------
__global__ void __launch_bounds__(128)
argmin_kernel(const float* __restrict__ x,          // [32][256][1024] NCHW
              const float* __restrict__ embed,      // [256][1024] f32
              const _Float16* __restrict__ embedT,  // [1024][256] f16
              const float* __restrict__ norms,      // [1024]
              float* __restrict__ out,              // [32][256][1024] NCHW
              float* __restrict__ partial,          // [512] block diff sums
              int* __restrict__ counts) {           // [1024]
    const int blk     = blockIdx.x;
    const int b       = blk >> 4;
    const int pixBase = (blk & 15) * MTILE;
    const int tid     = threadIdx.x;
    const int wave    = tid >> 5;
    const int lane    = tid & 31;
    const int laneLo  = lane & 15;
    const int half    = lane >> 4;

    __shared__ _Float16 ldsA[MTILE * ASTRIDE];                 // 33,792 B
    __shared__ __align__(16) _Float16 ldsB[2][16 * DIMC];      // 16,384 B
    __shared__ float ldsNorm[NE];                              //  4,096 B
    __shared__ float xnPart[128];
    __shared__ float xnorm[MTILE];
    __shared__ float diffTile[MTILE];
    __shared__ int   idxTile[MTILE];
    __shared__ float ls2[4];

    // ---- Stage x tile into LDS: [c][p] -> [p][c], f32 -> f16 (coalesced).
    //      p = tid & 63 is loop-invariant: accumulate ||x_p||^2 in f32 free.
    const float* xb = x + (size_t)b * DIMC * HW + pixBase;
    float xn = 0.f;
    for (int it = 0; it < 128; ++it) {
        int id = it * 128 + tid;
        int c  = id >> 6;
        int p  = id & 63;
        float v = xb[(size_t)c * HW + p];
        xn += v * v;
        ldsA[p * ASTRIDE + c] = (_Float16)v;
    }
    xnPart[tid] = xn;
    // ---- Stage code norms into LDS (coalesced) ----
    for (int r = 0; r < 8; ++r) ldsNorm[r * 128 + tid] = norms[r * 128 + tid];
    // ---- Stage B tile 0 into LDS buffer 0 (8KB cooperative, 16B/lane) ----
    {
        const uint4* src = (const uint4*)embedT;
        uint4*       dst = (uint4*)&ldsB[0][0];
        for (int r = 0; r < 4; ++r) dst[r * 128 + tid] = src[r * 128 + tid];
    }
    __syncthreads();
    if (tid < MTILE) xnorm[tid] = xnPart[tid] + xnPart[tid + 64];

    // ---- Load this wave's entire A block (16 rows x 256 K): 8 fragments --
    // 16-bit A 16x32 layout: lane(laneLo,half): K = half*8+e (e<8) and
    // 16+half*8+(e-8) (e>=8)  -> two aligned ds_load_b128 per fragment.
    v16h afrag[8];
    {
        const _Float16* arow = &ldsA[(wave * 16 + laneLo) * ASTRIDE];
        for (int kc = 0; kc < 8; ++kc) {
            union { v16h v; v8h h[2]; } u;
            u.h[0] = *(const v8h*)(arow + kc * 32 + half * 8);
            u.h[1] = *(const v8h*)(arow + kc * 32 + 16 + half * 8);
            afrag[kc] = u.v;
        }
    }

    // Running per-lane argmin: lane holds column slot laneLo of each code
    // tile, rows M = v + 8*half (C/D layout).
    float minV[8];
    int   minI[8];
    for (int v = 0; v < 8; ++v) { minV[v] = 3.4e38f; minI[v] = 0x7fffffff; }

    for (int tn = 0; tn < 64; ++tn) {
        const int cur = tn & 1;

        // -- Prefetch next B tile into registers (global loads issue early,
        //    latency overlaps the WMMA chain below) --
        uint4 pf[4];
        if (tn < 63) {
            const uint4* src = (const uint4*)(embedT + (size_t)(tn + 1) * 16 * DIMC);
            for (int r = 0; r < 4; ++r) pf[r] = src[r * 128 + tid];
        }

        // -- B fragments from LDS: 32x16 layout, lane(n=laneLo, half):
        //    K = half*8+e (e<8) and 16+half*8+(e-8) (e>=8) = 2x ds_load_b128 --
        const _Float16* brow = &ldsB[cur][laneLo * DIMC];
        v16h bf[8];
        for (int kc = 0; kc < 8; ++kc) {
            union { v16h v; v8h h[2]; } ub;
            ub.h[0] = *(const v8h*)(brow + kc * 32 + half * 8);
            ub.h[1] = *(const v8h*)(brow + kc * 32 + 16 + half * 8);
            bf[kc] = ub.v;
        }

        v8f acc = {};
        for (int kc = 0; kc < 8; ++kc)
            acc = __builtin_amdgcn_wmma_f32_16x16x32_f16(
                false, afrag[kc], false, bf[kc], (short)0, acc, false, false);

        const int   j  = tn * 16 + laneLo;       // this lane's code column
        const float nj = ldsNorm[j];
        for (int v = 0; v < 8; ++v) {
            float s = nj - 2.0f * acc[v];        // ||x||^2 per-row const: drop
            if (s < minV[v]) { minV[v] = s; minI[v] = j; }   // j ascending
        }

        // -- Spill prefetched tile into the other LDS buffer --
        if (tn < 63) {
            uint4* dst = (uint4*)&ldsB[cur ^ 1][0];
            for (int r = 0; r < 4; ++r) dst[r * 128 + tid] = pf[r];
        }
        __syncthreads();
    }

    // Reduce across the 16 column lanes (per half), first-min tie-break.
    for (int m = 1; m <= 8; m <<= 1) {
        for (int v = 0; v < 8; ++v) {
            float ov = __shfl_xor(minV[v], m, 32);
            int   oi = __shfl_xor(minI[v], m, 32);
            if (ov < minV[v] || (ov == minV[v] && oi < minI[v])) {
                minV[v] = ov; minI[v] = oi;
            }
        }
    }
    if (laneLo == 0) {
        for (int v = 0; v < 8; ++v) {
            int row = wave * 16 + half * 8 + v;
            idxTile[row]  = minI[v];
            diffTile[row] = minV[v];
            atomicAdd(&counts[minI[v]], 1);
        }
    }
    __syncthreads();

    // ---- Per-block commitment-loss partial: sum(xnorm[p] + minV[p]) ----
    {
        float d = (tid < MTILE) ? (xnorm[tid] + diffTile[tid]) : 0.f;
        for (int m = 16; m >= 1; m >>= 1) d += __shfl_xor(d, m, 32);
        if (lane == 0) ls2[wave] = d;
        __syncthreads();
        if (tid == 0) partial[blk] = ls2[0] + ls2[1] + ls2[2] + ls2[3];
    }

    // ---- Gather + write output from the original f32 embed ----
    // Thread owns pixel p = tid&63; iterates channels (2 per step).
    {
        const int p = tid & 63;
        const int j = idxTile[p];
        float* outb = out + (size_t)b * DIMC * HW + pixBase;
        for (int it = 0; it < 128; ++it) {
            int c = it * 2 + (tid >> 6);
            outb[(size_t)c * HW + p] = embed[(size_t)c * NE + j];
        }
    }
}

// ---------------------------------------------------------------------------
// Kernel 3: reduce 16 tile-partials per image -> diff[b] (32 blocks, 1 wave).
// ---------------------------------------------------------------------------
__global__ void __launch_bounds__(32)
finalize_diff_kernel(const float* __restrict__ partial,   // [512]
                     float* __restrict__ tail) {          // tail[0..31] = diff
    const int b = blockIdx.x;
    float s = (threadIdx.x < 16) ? partial[b * 16 + threadIdx.x] : 0.f;
    for (int m = 16; m >= 1; m >>= 1) s += __shfl_xor(s, m, 32);
    if (threadIdx.x == 0)
        tail[b] = s * (1.0f / ((float)HW * (float)DIMC));
}

// ---------------------------------------------------------------------------
// Kernel 4: perplexity from histogram (1 block).
// ---------------------------------------------------------------------------
__global__ void __launch_bounds__(256)
finalize_perp_kernel(const int* __restrict__ counts,
                     float* __restrict__ tail) {   // tail[32] = perplexity
    float s = 0.f;
    for (int r = 0; r < 4; ++r) {
        int   j = r * 256 + threadIdx.x;
        float p = (float)counts[j] * (1.0f / (float)NPIX);
        s += p * logf(p + 1e-10f);
    }
    for (int m = 16; m >= 1; m >>= 1) s += __shfl_xor(s, m, 32);
    __shared__ float ls[8];
    const int wave = threadIdx.x >> 5, lane = threadIdx.x & 31;
    if (lane == 0) ls[wave] = s;
    __syncthreads();
    if (threadIdx.x == 0) {
        float t = 0.f;
        for (int w = 0; w < 8; ++w) t += ls[w];
        tail[32] = expf(-t);
    }
}

// ---------------------------------------------------------------------------
// Workspace layout (bytes):
//   [0,       524288)  embedT  f16 [1024][256]
//   [524288,  528384)  norms   f32 [1024]
//   [528384,  532480)  counts  i32 [1024]
//   [532480,  534528)  partial f32 [512]
// ---------------------------------------------------------------------------
extern "C" void kernel_launch(void* const* d_in, const int* in_sizes, int n_in,
                              void* d_out, int out_size, void* d_ws, size_t ws_size,
                              hipStream_t stream) {
    const float* x     = (const float*)d_in[0];   // [32,256,32,32]
    const float* embed = (const float*)d_in[1];   // [256,1024]
    float* out  = (float*)d_out;
    float* tail = out + (size_t)OUTN;             // diff[32] then perplexity

    char* ws = (char*)d_ws;
    _Float16* embedT = (_Float16*)(ws);
    float*    norms  = (float*)(ws + 524288);
    int*      counts = (int*)(ws + 528384);
    float*    partial= (float*)(ws + 532480);

    hipLaunchKernelGGL(prep_kernel, dim3(NE), dim3(256), 0, stream,
                       embed, embedT, norms, counts);
    hipLaunchKernelGGL(argmin_kernel, dim3(NPIX / MTILE), dim3(128), 0, stream,
                       x, embed, embedT, norms, out, partial, counts);
    hipLaunchKernelGGL(finalize_diff_kernel, dim3(NB), dim3(32), 0, stream,
                       partial, tail);
    hipLaunchKernelGGL(finalize_perp_kernel, dim3(1), dim3(256), 0, stream,
                       counts, tail);
}
